// DecoderRNN_24584392803120
// MI455X (gfx1250) — compile-verified, matrix operand order checked by reading, not run on hardware
//
#include <hip/hip_runtime.h>
#include <hip/hip_bf16.h>
#include <math.h>

// ---------------- problem constants ----------------
#define Bm 128
#define Tm 512
#define Vm 1024
#define Em 256
#define Hm 512

typedef _Float16 f16;
typedef __attribute__((ext_vector_type(16))) _Float16 v16h;
typedef __attribute__((ext_vector_type(8)))  _Float16 v8h;
typedef __attribute__((ext_vector_type(8)))  float    v8f;

// gfx1250 async global->LDS path (ASYNCcnt-tracked), with portable fallback
#if defined(__has_builtin)
#if __has_builtin(__builtin_amdgcn_global_load_async_to_lds_b128) && \
    __has_builtin(__builtin_amdgcn_s_wait_asynccnt)
#define USE_ASYNC_LDS 1
#endif
#endif

#ifdef USE_ASYNC_LDS
// toolchain-probed signature: (int __vector(4)*, AS3 int __vector(4)*, imm, imm)
typedef int v4i_gcc __attribute__((vector_size(16)));
typedef __attribute__((address_space(3))) v4i_gcc lds_v4i;

__device__ __forceinline__ void async_copy16(const f16* g, f16* l) {
  __builtin_amdgcn_global_load_async_to_lds_b128(
      (v4i_gcc*)const_cast<f16*>(g), (lds_v4i*)l, 0, 0);
}
#endif

// -----------------------------------------------------------------------------
// WMMA fragment loader.
// CDNA5 16-bit A-matrix (16x32, MxK) layout (ISA 7.12.2):
//   lane<16:  m=lane,    elems 0..7 -> K=kb+0..7,  elems 8..15 -> K=kb+16..23
//   lane>=16: m=lane-16, elems 0..7 -> K=kb+8..15, elems 8..15 -> K=kb+24..31
// Each lane: two contiguous 16B chunks at kb+8*hi and kb+16+8*hi.
// B-matrix (KxN) mirrors this with n = lane%16 and B[k][n] = W[n][k] for
// row-major W (computing X @ W^T), so the SAME loader serves B fragments.
// -----------------------------------------------------------------------------
__device__ __forceinline__ v16h load_frag_rm(const f16* __restrict__ base,
                                             int stride_halfs, int kb) {
  const int lane = threadIdx.x & 31;
  const int r    = lane & 15;
  const int hi   = lane >> 4;
  const f16* p = base + (size_t)r * stride_halfs + kb + (hi << 3);
  v8h c0 = *(const v8h*)(p);
  v8h c1 = *(const v8h*)(p + 16);
  v16h out;
#pragma unroll
  for (int i = 0; i < 8; ++i) { out[i] = c0[i]; out[i + 8] = c1[i]; }
  return out;
}

__device__ __forceinline__ float sigmoidf_(float x) {
  return 1.0f / (1.0f + __expf(-x));
}

// software grid barrier (persistent cooperative kernel, 32 resident WGs)
__device__ __forceinline__ void grid_sync(unsigned* cnt, unsigned* gen, unsigned nwg) {
  __threadfence();
  __syncthreads();
  if (threadIdx.x == 0) {
    unsigned g = __hip_atomic_load(gen, __ATOMIC_ACQUIRE, __HIP_MEMORY_SCOPE_AGENT);
    unsigned arrived = __hip_atomic_fetch_add(cnt, 1u, __ATOMIC_ACQ_REL, __HIP_MEMORY_SCOPE_AGENT);
    if (arrived == nwg - 1) {
      __hip_atomic_store(cnt, 0u, __ATOMIC_RELAXED, __HIP_MEMORY_SCOPE_AGENT);
      __hip_atomic_fetch_add(gen, 1u, __ATOMIC_ACQ_REL, __HIP_MEMORY_SCOPE_AGENT);
    } else {
      while (__hip_atomic_load(gen, __ATOMIC_ACQUIRE, __HIP_MEMORY_SCOPE_AGENT) == g)
        __builtin_amdgcn_s_sleep(1);
    }
  }
  __syncthreads();
}

// ---------------- kernel 1: W_out -> f16, init barrier words ----------------
__global__ void prep_kernel(const float* __restrict__ Wout, f16* __restrict__ Wout16,
                            unsigned* __restrict__ bar) {
  int idx = blockIdx.x * blockDim.x + threadIdx.x;
  if (idx < 2) bar[idx] = 0u;
  if (idx < Vm * Hm) Wout16[idx] = (f16)Wout[idx];
}

// ---------------- kernel 2: embed + relu, time-major f16 X ----------------
__global__ void embed_kernel(const int* __restrict__ target, const float* __restrict__ emb,
                             f16* __restrict__ X) {
  int idx = blockIdx.x * blockDim.x + threadIdx.x;      // ((t*B)+b)*E + e
  if (idx >= Tm * Bm * Em) return;
  int e = idx % Em;
  int b = (idx / Em) % Bm;
  int t = idx / (Em * Bm);
  int tok = (t == 0) ? 0 : target[(size_t)b * Tm + (t - 1)];
  float v = emb[(size_t)tok * Em + e];
  X[idx] = (f16)(v > 0.f ? v : 0.f);
}

// ---------------- kernel 3: persistent cooperative LSTM scan ----------------
// grid = 32 WGs x 256 threads (8 wave32s). WG jt owns hidden cols [16*jt,16*jt+16)
// and therefore gate cols {g*512+16*jt .. +16} for g=i,f,g,o -> local c/h update.
// Wave w owns batch rows [16w, 16w+16). Weights staged once in 96KB dynamic LDS.
// Persistent (exactly 32 WGs) -> occupancy is irrelevant; give the compiler the
// full VGPR budget so loop addresses/fragments never spill to scratch.
__global__ void __launch_bounds__(256, 1)
lstm_scan_kernel(const f16* __restrict__ X,        // (T,B,E) f16
                 const float* __restrict__ enc_h,  // (2,B,H)
                 const float* __restrict__ enc_c,  // (2,B,H)
                 const float* __restrict__ W_ih,   // (4H,E)
                 const float* __restrict__ W_hh,   // (4H,H)
                 const float* __restrict__ b_ih,
                 const float* __restrict__ b_hh,
                 f16* __restrict__ hbuf,           // 2*(B,H) double buffer
                 f16* __restrict__ hs,             // (B,T,H) f16
                 float* __restrict__ outh,         // (B,H) f32
                 float* __restrict__ outc,         // (B,H) f32
                 unsigned* __restrict__ cnt, unsigned* __restrict__ gen) {
  extern __shared__ f16 smem[];
  f16* Wih_s = smem;                  // [4][16][E]  32KB
  f16* Whh_s = smem + 4 * 16 * Em;    // [4][16][H]  64KB
  const int jt = blockIdx.x;
  const int j0 = jt * 16;
  const int tid = threadIdx.x;

  // stage this WG's gate-column weights into LDS (f32 -> f16)
  for (int idx = tid; idx < 4 * 16 * Em; idx += 256) {
    int k = idx % Em; int n16 = (idx / Em) & 15; int g = idx / (16 * Em);
    Wih_s[idx] = (f16)W_ih[(size_t)(g * Hm + j0 + n16) * Em + k];
  }
  for (int idx = tid; idx < 4 * 16 * Hm; idx += 256) {
    int k = idx % Hm; int n16 = (idx / Hm) & 15; int g = idx / (16 * Hm);
    Whh_s[idx] = (f16)W_hh[(size_t)(g * Hm + j0 + n16) * Hm + k];
  }
  // h0 = mean over layer dim, this WG's 16 columns, all batches -> hbuf[0]
  for (int idx = tid; idx < Bm * 16; idx += 256) {
    int b = idx >> 4; int j = j0 + (idx & 15);
    float h0 = 0.5f * (enc_h[(size_t)b * Hm + j] + enc_h[(size_t)(Bm + b) * Hm + j]);
    hbuf[(size_t)b * Hm + j] = (f16)h0;
  }
  __syncthreads();

  const int lane = tid & 31, w = tid >> 5;
  const int nl = lane & 15, hi = lane >> 4;

  float biasv[4];
#pragma unroll
  for (int g = 0; g < 4; ++g)
    biasv[g] = b_ih[g * Hm + j0 + nl] + b_hh[g * Hm + j0 + nl];

  // c0 fragment (C-layout: elem r, lane -> row 16w + r + 8*hi, col j0+nl)
  v8f c;
#pragma unroll
  for (int r = 0; r < 8; ++r) {
    int b = 16 * w + r + 8 * hi;
    c[r] = 0.5f * (enc_c[(size_t)b * Hm + j0 + nl] + enc_c[(size_t)(Bm + b) * Hm + j0 + nl]);
  }

  grid_sync(cnt, gen, gridDim.x);  // h0 visible device-wide

  for (int t = 0; t < Tm; ++t) {
    const f16* hcur = hbuf + (size_t)(t & 1) * (Bm * Hm);
    f16* hnxt       = hbuf + (size_t)((t + 1) & 1) * (Bm * Hm);

    v8f acc[4];
#pragma unroll
    for (int g = 0; g < 4; ++g)
#pragma unroll
      for (int r = 0; r < 8; ++r) acc[g][r] = biasv[g];

    // x-side: gates += x_t @ W_ih^T   (K = 256)
    const f16* xa = X + (size_t)t * (Bm * Em) + (size_t)(16 * w) * Em;
    // prefetch next timestep's x tile behind this step's WMMAs (global_prefetch_b8)
    if (t + 1 < Tm)
      __builtin_prefetch(xa + (size_t)Bm * Em + (size_t)(tid & 15) * Em, 0, 3);
    for (int kb = 0; kb < Em; kb += 32) {
      v16h a = load_frag_rm(xa, Em, kb);
#pragma unroll
      for (int g = 0; g < 4; ++g) {
        v16h bf = load_frag_rm(Wih_s + g * 16 * Em, Em, kb);
        acc[g] = __builtin_amdgcn_wmma_f32_16x16x32_f16(false, a, false, bf,
                                                        (short)0, acc[g], false, false);
      }
    }
    // h-side: gates += h @ W_hh^T     (K = 512)
    const f16* ha = hcur + (size_t)(16 * w) * Hm;
    for (int kb = 0; kb < Hm; kb += 32) {
      v16h a = load_frag_rm(ha, Hm, kb);
#pragma unroll
      for (int g = 0; g < 4; ++g) {
        v16h bf = load_frag_rm(Whh_s + g * 16 * Hm, Hm, kb);
        acc[g] = __builtin_amdgcn_wmma_f32_16x16x32_f16(false, a, false, bf,
                                                        (short)0, acc[g], false, false);
      }
    }

    // elementwise LSTM cell (torch gate order i,f,g,o), fully local to this lane
#pragma unroll
    for (int r = 0; r < 8; ++r) {
      float ig = sigmoidf_(acc[0][r]);
      float fg = sigmoidf_(acc[1][r]);
      float gg = tanhf(acc[2][r]);
      float og = sigmoidf_(acc[3][r]);
      float cn = fg * c[r] + ig * gg;
      float hn = og * tanhf(cn);
      c[r] = cn;
      int b = 16 * w + r + 8 * hi;
      int j = j0 + nl;
      hnxt[(size_t)b * Hm + j] = (f16)hn;
      hs[((size_t)b * Tm + t) * Hm + j] = (f16)hn;
      if (t == Tm - 1) {
        outh[(size_t)b * Hm + j] = hn;
        outc[(size_t)b * Hm + j] = cn;
      }
    }
    grid_sync(cnt, gen, gridDim.x);
  }
}

// ---------------- kernel 4: logits GEMM + exact log_softmax ----------------
// one WG per 16 rows of (B*T, V); wave w owns cols [128w, 128w+128) (8 tiles)
__global__ void __launch_bounds__(256, 1)
out_logsoftmax_kernel(const f16* __restrict__ hs,    // (B*T, H)
                      const f16* __restrict__ Wout,  // (V, H) f16
                      const float* __restrict__ b_out,
                      float* __restrict__ outp) {
  __shared__ __align__(16) f16 atile[16 * Hm];   // 16KB, contiguous hs range
  __shared__ float red[2][8][16];
  const int row0 = blockIdx.x * 16;
  const int tid = threadIdx.x;

  // stage 16 contiguous hs rows (16KB) into LDS
  {
    const f16* src = hs + (size_t)row0 * Hm;
#ifdef USE_ASYNC_LDS
    // 1024 x 16B chunks, ASYNCcnt-tracked async copy to LDS
    for (int idx = tid; idx < (16 * Hm) / 8; idx += 256)
      async_copy16(src + (size_t)idx * 8, atile + (size_t)idx * 8);
    __builtin_amdgcn_s_wait_asynccnt(0);
#else
    for (int idx = tid; idx < (16 * Hm) / 8; idx += 256)
      ((v8h*)atile)[idx] = ((const v8h*)src)[idx];
#endif
  }
  __syncthreads();

  const int lane = tid & 31, w = tid >> 5;
  const int nl = lane & 15, hi = lane >> 4;

  v8f acc[8];
#pragma unroll
  for (int tc = 0; tc < 8; ++tc) {
    float bz = b_out[w * 128 + tc * 16 + nl];
#pragma unroll
    for (int r = 0; r < 8; ++r) acc[tc][r] = bz;
  }
  for (int kb = 0; kb < Hm; kb += 32) {
    v16h a = load_frag_rm(atile, Hm, kb);
#pragma unroll
    for (int tc = 0; tc < 8; ++tc) {
      v16h bf = load_frag_rm(Wout + (size_t)(w * 128 + tc * 16) * Hm, Hm, kb);
      acc[tc] = __builtin_amdgcn_wmma_f32_16x16x32_f16(false, a, false, bf,
                                                       (short)0, acc[tc], false, false);
    }
  }

  // row max over this wave's 128 cols (lanes 0-15 carry row r, 16-31 row r+8)
  float rowmax[8];
#pragma unroll
  for (int r = 0; r < 8; ++r) {
    float m = acc[0][r];
#pragma unroll
    for (int tc = 1; tc < 8; ++tc) m = fmaxf(m, acc[tc][r]);
    for (int s = 1; s <= 8; s <<= 1) m = fmaxf(m, __shfl_xor(m, s, 32));
    if (nl == 0) red[0][w][r + 8 * hi] = m;
  }
  __syncthreads();
#pragma unroll
  for (int r = 0; r < 8; ++r) {
    float m = red[0][0][r + 8 * hi];
#pragma unroll
    for (int ww = 1; ww < 8; ++ww) m = fmaxf(m, red[0][ww][r + 8 * hi]);
    rowmax[r] = m;
  }
  // sum of exp
  float rlog[8];
#pragma unroll
  for (int r = 0; r < 8; ++r) {
    float s = 0.f;
#pragma unroll
    for (int tc = 0; tc < 8; ++tc) s += __expf(acc[tc][r] - rowmax[r]);
    for (int sh = 1; sh <= 8; sh <<= 1) s += __shfl_xor(s, sh, 32);
    if (nl == 0) red[1][w][r + 8 * hi] = s;
  }
  __syncthreads();
#pragma unroll
  for (int r = 0; r < 8; ++r) {
    float s = 0.f;
#pragma unroll
    for (int ww = 0; ww < 8; ++ww) s += red[1][ww][r + 8 * hi];
    rlog[r] = rowmax[r] + __logf(s);
  }
  // write log-probs
#pragma unroll
  for (int tc = 0; tc < 8; ++tc)
#pragma unroll
    for (int r = 0; r < 8; ++r)
      outp[(size_t)(row0 + r + 8 * hi) * Vm + (w * 128 + tc * 16 + nl)] =
          acc[tc][r] - rlog[r];
}

// -----------------------------------------------------------------------------
extern "C" void kernel_launch(void* const* d_in, const int* in_sizes, int n_in,
                              void* d_out, int out_size, void* d_ws, size_t ws_size,
                              hipStream_t stream) {
  (void)in_sizes; (void)n_in; (void)out_size; (void)ws_size;
  // inputs per setup_inputs() order
  const float* enc_h  = (const float*)d_in[1];
  const float* enc_c  = (const float*)d_in[2];
  const int*   target = (const int*)d_in[3];
  const float* emb    = (const float*)d_in[4];
  const float* W_ih   = (const float*)d_in[5];
  const float* W_hh   = (const float*)d_in[6];
  const float* b_ih   = (const float*)d_in[7];
  const float* b_hh   = (const float*)d_in[8];
  const float* W_out  = (const float*)d_in[9];
  const float* b_out  = (const float*)d_in[10];

  // workspace layout (bytes)
  char* ws = (char*)d_ws;
  f16* X       = (f16*)(ws);                              // 33,554,432 B
  f16* hs      = (f16*)(ws + 33554432);                   // 67,108,864 B
  f16* hbuf    = (f16*)(ws + 100663296);                  //    262,144 B
  f16* Wout16  = (f16*)(ws + 100925440);                  //  1,048,576 B
  unsigned* bar = (unsigned*)(ws + 101974016);            //          8 B

  float* outp = (float*)d_out;                 // (B,T,V) log-probs
  float* outh = outp + (size_t)Bm * Tm * Vm;   // (1,B,H) h_T
  float* outc = outh + (size_t)Bm * Hm;        // (1,B,H) c_T

  prep_kernel<<<(Vm * Hm + 255) / 256, 256, 0, stream>>>(W_out, Wout16, bar);
  embed_kernel<<<(Tm * Bm * Em + 255) / 256, 256, 0, stream>>>(target, emb, X);
  lstm_scan_kernel<<<32, 256, 96 * 1024, stream>>>(X, enc_h, enc_c, W_ih, W_hh,
                                                   b_ih, b_hh, hbuf, hs,
                                                   outh, outc, bar + 0, bar + 1);
  out_logsoftmax_kernel<<<(Bm * Tm) / 16, 256, 0, stream>>>(hs, Wout16, b_out, outp);
}
